// FuXiBlock_38929583571112
// MI455X (gfx1250) — compile-verified, matrix operand order checked by reading, not run on hardware
//
#include <hip/hip_runtime.h>
#include <hip/hip_bf16.h>
#include <math.h>

// ---------------------------------------------------------------------------
// FuXi block for MI455X (gfx1250, wave32, WMMA).
// bf16 operands + fp32 accumulate via v_wmma_f32_16x16x32_bf16:
// ~103 GFLOP at the bf16 matrix rate (~20us) sits at the 23.3 TB/s HBM
// roofline (~17us for ~0.4GB, dominated by p_bias); fp32 WMMA (K=4) would be
// ~8x slower. Element-wise math stays fp32.
// GEMM blocks stage weight tiles in LDS via GLOBAL_LOAD_ASYNC_TO_LDS (ASYNCcnt
// path) so 8 waves share one fetch; WMMA B-fragments then come from ds_load.
// ---------------------------------------------------------------------------

typedef __bf16 bf16_t;
typedef __attribute__((ext_vector_type(16))) __bf16 v16bf;
typedef __attribute__((ext_vector_type(8)))  __bf16 v8bf;
typedef __attribute__((ext_vector_type(8)))  float  v8f;
typedef int v4i_ __attribute__((vector_size(16)));   // matches builtin param type

#define AS3 __attribute__((address_space(3)))

#if __has_builtin(__builtin_amdgcn_global_load_async_to_lds_b128)
#define HAVE_ASYNC_LDS 1
#else
#define HAVE_ASYNC_LDS 0
#endif

__device__ __forceinline__ void wait_async0() {
#if __has_builtin(__builtin_amdgcn_s_wait_asynccnt)
  __builtin_amdgcn_s_wait_asynccnt(0);
#else
  asm volatile("s_wait_asynccnt 0x0" ::: "memory");
#endif
}

__device__ __forceinline__ float silu_f(float x) {
  return x / (1.0f + __expf(-x));
}

// Load a 16x32 bf16 A/B-style fragment (ISA 7.12.2) from row-major [16][ld].
// Lane L<16: row=L, K in {0..7, 16..23}; lane L>=16: row=L-16, K {8..15,24..31}.
__device__ __forceinline__ v16bf load_frag(const bf16_t* __restrict__ base, int ld) {
  const int lane = threadIdx.x & 31;
  const int row  = lane & 15;
  const int kb   = (lane & 16) ? 8 : 0;
  const bf16_t* p = base + (size_t)row * (size_t)ld + kb;
  v8bf lo = *(const v8bf*)(p);
  v8bf hi = *(const v8bf*)(p + 16);
  v16bf r;
#pragma unroll
  for (int i = 0; i < 8; ++i) { r[i] = lo[i]; r[i + 8] = hi[i]; }
  return r;
}

// Same fragment pattern out of an LDS tile with fixed row stride 32.
__device__ __forceinline__ v16bf load_frag_lds(const bf16_t* tile /*[16][32]*/) {
  const int lane = threadIdx.x & 31;
  const int row  = lane & 15;
  const int kb   = (lane & 16) ? 8 : 0;
  const bf16_t* p = tile + row * 32 + kb;
  v8bf lo = *(const v8bf*)(p);
  v8bf hi = *(const v8bf*)(p + 16);
  v16bf r;
#pragma unroll
  for (int i = 0; i < 8; ++i) { r[i] = lo[i]; r[i + 8] = hi[i]; }
  return r;
}

// ---------------------------------------------------------------------------
// fp32 -> bf16 cast
// ---------------------------------------------------------------------------
__global__ void k_cast(const float* __restrict__ a, bf16_t* __restrict__ o, int n) {
  int i = blockIdx.x * blockDim.x + threadIdx.x;
  if (i < n) o[i] = (bf16_t)a[i];
}

// ---------------------------------------------------------------------------
// RMSNorm: one block per row, fp32 in, bf16 out
// ---------------------------------------------------------------------------
__global__ void k_rmsnorm(const float* __restrict__ x, const float* __restrict__ g,
                          bf16_t* __restrict__ out, int D) {
  __shared__ float red[256];
  const int row = blockIdx.x;
  const float* xr = x + (size_t)row * D;
  float s = 0.f;
  for (int i = threadIdx.x; i < D; i += 256) { float v = xr[i]; s += v * v; }
  red[threadIdx.x] = s;
  __syncthreads();
  for (int off = 128; off > 0; off >>= 1) {
    if ((int)threadIdx.x < off) red[threadIdx.x] += red[threadIdx.x + off];
    __syncthreads();
  }
  const float inv = rsqrtf(red[0] / (float)D + 1e-8f);
  for (int i = threadIdx.x; i < D; i += 256)
    out[(size_t)row * D + i] = (bf16_t)(g[i] * (xr[i] * inv));
}

// ---------------------------------------------------------------------------
// WMMA GEMM: C[M][N] = epi(A[M][K] @ W[N][K]^T).
// Block = 256 threads = 8 waves = 128M x 64N block tile; each wave owns a
// 16x64 strip (4 accumulators). The 64x32 bf16 weight tile for each K-step is
// staged once per block into LDS with async global->LDS copies (one b128 per
// thread), then all 8 waves read their B fragments via ds_load. A fragments
// come straight from global (activations live in the 192MB L2).
// EPI: 0 = store bf16; 1 = silu -> fp32; 2 = + residual -> fp32; 3 = raw fp32
// ---------------------------------------------------------------------------
template <int EPI>
__global__ void k_gemm(const bf16_t* __restrict__ A, int lda,
                       const bf16_t* __restrict__ W, int ldw,
                       void* __restrict__ outp, int ldo,
                       const float* __restrict__ res,
                       int M, int N, int K) {
  __shared__ bf16_t wt[64][32];                  // 4KB weight stage
  const int wave = threadIdx.x >> 5;
  const int lane = threadIdx.x & 31;
  const int tiles_n = N >> 6;
  const int bm = (blockIdx.x / tiles_n) << 7;    // block row base (128)
  const int tn = (blockIdx.x % tiles_n) << 6;    // block col base (64)
  const int tm = bm + (wave << 4);               // this wave's 16-row strip

  // weight-stage assignment: thread t -> row t/4, 8-element segment t%4
  const int wrow = threadIdx.x >> 2;
  const int wseg = (threadIdx.x & 3) << 3;

  v8f acc[4] = {};
  for (int k = 0; k < K; k += 32) {
    const bf16_t* wsrc = W + (size_t)(tn + wrow) * ldw + k + wseg;
#if HAVE_ASYNC_LDS
    __builtin_amdgcn_global_load_async_to_lds_b128(
        (v4i_*)wsrc, (AS3 v4i_*)&wt[wrow][wseg], 0, 0);
    wait_async0();
#else
    *(v8bf*)&wt[wrow][wseg] = *(const v8bf*)wsrc;
#endif
    __syncthreads();

    v16bf a = load_frag(A + (size_t)tm * lda + k, lda);
#pragma unroll
    for (int j = 0; j < 4; ++j) {
      v16bf b = load_frag_lds(&wt[j * 16][0]);
      acc[j] = __builtin_amdgcn_wmma_f32_16x16x32_bf16(
          false, a, false, b, (short)0, acc[j], false, false);
    }
    __syncthreads();
  }

  const int hi  = (lane & 16) ? 8 : 0;
  const int col = lane & 15;
#pragma unroll
  for (int j = 0; j < 4; ++j) {
#pragma unroll
    for (int r = 0; r < 8; ++r) {
      const int m = tm + r + hi;
      const int n = tn + j * 16 + col;
      const float v = acc[j][r];
      if (EPI == 0)      ((bf16_t*)outp)[(size_t)m * ldo + n] = (bf16_t)v;
      else if (EPI == 1) ((float*)outp)[(size_t)m * ldo + n] = silu_f(v);
      else if (EPI == 2) ((float*)outp)[(size_t)m * ldo + n] = v + res[(size_t)m * ldo + n];
      else               ((float*)outp)[(size_t)m * ldo + n] = v;
    }
  }
}

// ---------------------------------------------------------------------------
// Transpose V: vT[b*H+h][dh][s] = qkv[b*S+s][2][h][dh]  (bf16)
// ---------------------------------------------------------------------------
__global__ void k_vT(const bf16_t* __restrict__ qkv, bf16_t* __restrict__ vT,
                     int B, int S, int H) {
  const int Dh = 64, ld = 3072, D = 1024;
  int i = blockIdx.x * blockDim.x + threadIdx.x;
  int total = B * H * Dh * S;
  if (i >= total) return;
  int s  = i % S;
  int dh = (i / S) % Dh;
  int bh = i / (S * Dh);
  int b = bh / H, h = bh % H;
  vT[i] = qkv[((size_t)(b * S + s)) * ld + 2 * D + h * Dh + dh];
}

// ---------------------------------------------------------------------------
// Attention: per-wave 16-row q-tile, 32-token causal k-chunks.
// scores: 4 WMMAs; silu(scale*S + p_bias + mask) -> per-wave LDS -> A-frag;
// P@V: 4 WMMAs against vT. fp32 accumulation.
// ---------------------------------------------------------------------------
__global__ void k_attn(const bf16_t* __restrict__ qkv, const bf16_t* __restrict__ vT,
                       const float* __restrict__ p_bias, float* __restrict__ attn,
                       int B, int S, int H) {
  const int D = 1024, Dh = 64, ld = 3072;
  __shared__ bf16_t psc[8][16][32];              // per-wave P scratch
  const int wave = threadIdx.x >> 5;
  const int lane = threadIdx.x & 31;
  const int qt_per = S >> 4;
  const int task = blockIdx.x * 8 + wave;
  if (task >= B * H * qt_per) return;
  const int qt = task % qt_per;
  const int bh = task / qt_per;
  const int b = bh / H, h = bh % H;
  const int q0 = qt << 4;
  const size_t baseTok = (size_t)b * S;

  const bf16_t* qbase = qkv + (baseTok + q0) * ld + h * Dh;
  const v16bf qf0 = load_frag(qbase, ld);        // dh 0..31
  const v16bf qf1 = load_frag(qbase + 32, ld);   // dh 32..63

  v8f acc[4] = {};
  const int hi  = (lane & 16) ? 8 : 0;
  const int col = lane & 15;
  const float scale = 0.125f;                    // Dh^-0.5

  for (int kc = 0; kc < q0 + 16; kc += 32) {
    const bf16_t* kb0 = qkv + (baseTok + kc) * ld + D + h * Dh;
    const bf16_t* kb1 = qkv + (baseTok + kc + 16) * ld + D + h * Dh;
    v8f s0 = {}, s1 = {};
    {
      v16bf f = load_frag(kb0, ld);
      s0 = __builtin_amdgcn_wmma_f32_16x16x32_bf16(false, qf0, false, f, (short)0, s0, false, false);
      f = load_frag(kb0 + 32, ld);
      s0 = __builtin_amdgcn_wmma_f32_16x16x32_bf16(false, qf1, false, f, (short)0, s0, false, false);
      f = load_frag(kb1, ld);
      s1 = __builtin_amdgcn_wmma_f32_16x16x32_bf16(false, qf0, false, f, (short)0, s1, false, false);
      f = load_frag(kb1 + 32, ld);
      s1 = __builtin_amdgcn_wmma_f32_16x16x32_bf16(false, qf1, false, f, (short)0, s1, false, false);
    }
    // prefetch next chunk's bias rows into cache (global_prefetch_b8)
    if (kc + 32 < q0 + 16)
      __builtin_prefetch(&p_bias[((size_t)h * S + q0 + hi) * S + kc + 32 + col], 0, 0);

    // bias + analytic causal mask + silu -> LDS [row][k] bf16
#pragma unroll
    for (int t = 0; t < 2; ++t) {
#pragma unroll
      for (int r = 0; r < 8; ++r) {
        const int m  = r + hi;
        const int q  = q0 + m;
        const int kk = kc + t * 16 + col;
        float v = (t ? s1[r] : s0[r]) * scale;
        v += p_bias[((size_t)h * S + q) * S + kk];
        if (kk > q) v += -1.0e9f;
        psc[wave][m][t * 16 + col] = (bf16_t)silu_f(v);
      }
    }
    __builtin_amdgcn_wave_barrier();
    asm volatile("s_wait_dscnt 0x0" ::: "memory");   // cross-lane LDS visibility
    v16bf pf;
    {
      const int row = lane & 15;
      const int kbo = (lane & 16) ? 8 : 0;
#pragma unroll
      for (int i = 0; i < 8; ++i) {
        pf[i]     = psc[wave][row][kbo + i];
        pf[i + 8] = psc[wave][row][kbo + 16 + i];
      }
    }
    __builtin_amdgcn_wave_barrier();
    const bf16_t* vb = vT + ((size_t)bh * Dh) * S + kc;
#pragma unroll
    for (int j = 0; j < 4; ++j) {
      v16bf vf = load_frag(vb + (size_t)(j * 16) * S, S);
      acc[j] = __builtin_amdgcn_wmma_f32_16x16x32_bf16(
          false, pf, false, vf, (short)0, acc[j], false, false);
    }
  }

#pragma unroll
  for (int j = 0; j < 4; ++j)
#pragma unroll
    for (int r = 0; r < 8; ++r) {
      const int m = q0 + r + hi;
      const int n = h * Dh + j * 16 + col;
      attn[(baseTok + m) * (size_t)D + n] = acc[j][r];
    }
}

// ---------------------------------------------------------------------------
// Element-wise: gate multiply and SwiGLU
// ---------------------------------------------------------------------------
__global__ void k_ams(const float* __restrict__ attn, const float* __restrict__ u,
                      bf16_t* __restrict__ out, int n) {
  int i = blockIdx.x * blockDim.x + threadIdx.x;
  if (i < n) out[i] = (bf16_t)(attn[i] * u[i]);
}

__global__ void k_swiglu(const float* __restrict__ x12, bf16_t* __restrict__ out,
                         int M, int Dd) {
  int i = blockIdx.x * blockDim.x + threadIdx.x;
  if (i >= M * Dd) return;
  int m = i / Dd, j = i % Dd;
  float a = x12[(size_t)m * (2 * Dd) + j];
  float c = x12[(size_t)m * (2 * Dd) + Dd + j];
  out[i] = (bf16_t)(silu_f(a) * c);
}

// ---------------------------------------------------------------------------
// Launch
// ---------------------------------------------------------------------------
extern "C" void kernel_launch(void* const* d_in, const int* in_sizes, int n_in,
                              void* d_out, int out_size, void* d_ws, size_t ws_size,
                              hipStream_t stream) {
  (void)in_sizes; (void)n_in; (void)out_size; (void)ws_size;
  const int B = 2, S = 2048, D = 1024, H = 16;
  const int M = B * S;  // 4096 tokens

  const float* x      = (const float*)d_in[0];
  const float* p_bias = (const float*)d_in[1];
  /* mask d_in[2] applied analytically (tril / -1e9) */
  const float* w_qkv  = (const float*)d_in[3];
  const float* w_u    = (const float*)d_in[4];
  const float* g_ams  = (const float*)d_in[5];
  const float* w0     = (const float*)d_in[6];
  const float* w1     = (const float*)d_in[7];
  const float* w3     = (const float*)d_in[8];
  const float* g_mffn = (const float*)d_in[9];
  float* out = (float*)d_out;

  char* p = (char*)d_ws;
  auto wsalloc = [&](size_t bytes) {
    char* r = p;
    p += (bytes + 255) & ~(size_t)255;
    return (void*)r;
  };
  bf16_t* xn    = (bf16_t*)wsalloc((size_t)M * D * 2);
  bf16_t* wqkvb = (bf16_t*)wsalloc((size_t)3 * D * D * 2);
  bf16_t* wub   = (bf16_t*)wsalloc((size_t)D * D * 2);
  bf16_t* w0b   = (bf16_t*)wsalloc((size_t)D * D * 2);
  bf16_t* w1b   = (bf16_t*)wsalloc((size_t)2 * D * D * 2);
  bf16_t* w3b   = (bf16_t*)wsalloc((size_t)D * D * 2);
  bf16_t* qkvb  = (bf16_t*)wsalloc((size_t)M * 3 * D * 2);
  bf16_t* vTb   = (bf16_t*)wsalloc((size_t)B * H * 64 * S * 2);
  float*  usil  = (float*)wsalloc((size_t)M * D * 4);
  float*  attn  = (float*)wsalloc((size_t)M * D * 4);
  bf16_t* amsb  = (bf16_t*)wsalloc((size_t)M * D * 2);
  float*  o     = (float*)wsalloc((size_t)M * D * 4);
  bf16_t* onb   = (bf16_t*)wsalloc((size_t)M * D * 2);
  float*  x12   = (float*)wsalloc((size_t)M * 2 * D * 4);
  bf16_t* hid   = (bf16_t*)wsalloc((size_t)M * D * 2);

  auto cast = [&](const float* src, bf16_t* dst, int n) {
    k_cast<<<(n + 255) / 256, 256, 0, stream>>>(src, dst, n);
  };
  cast(w_qkv, wqkvb, 3 * D * D);
  cast(w_u,   wub,   D * D);
  cast(w0,    w0b,   D * D);
  cast(w1,    w1b,   2 * D * D);
  cast(w3,    w3b,   D * D);

  k_rmsnorm<<<M, 256, 0, stream>>>(x, g_ams, xn, D);

  auto gblocks = [&](int m, int n) { return (m >> 7) * (n >> 6); };

  // qkv = xn @ w_qkv^T                               (bf16 out)
  k_gemm<0><<<gblocks(M, 3 * D), 256, 0, stream>>>(xn, D, wqkvb, D, qkvb, 3 * D, nullptr, M, 3 * D, D);
  // vT for attention
  k_vT<<<(B * H * 64 * S + 255) / 256, 256, 0, stream>>>(qkvb, vTb, B, S, H);
  // u = silu(xn @ w_u^T)                             (fp32 out)
  k_gemm<1><<<gblocks(M, D), 256, 0, stream>>>(xn, D, wub, D, usil, D, nullptr, M, D, D);
  // attn = silu(scale*QK^T + bias + mask) @ V        (fp32 out)
  k_attn<<<(B * H * (S >> 4) + 7) / 8, 256, 0, stream>>>(qkvb, vTb, p_bias, attn, B, S, H);
  // ams = attn * u                                    (bf16 out)
  k_ams<<<(M * D + 255) / 256, 256, 0, stream>>>(attn, usil, amsb, M * D);
  // o = ams @ w0^T + x                                (fp32 out)
  k_gemm<2><<<gblocks(M, D), 256, 0, stream>>>(amsb, D, w0b, D, o, D, x, M, D, D);
  // on = rmsnorm(o)                                   (bf16 out)
  k_rmsnorm<<<M, 256, 0, stream>>>(o, g_mffn, onb, D);
  // x12 = on @ w1^T                                   (fp32 out)
  k_gemm<3><<<gblocks(M, 2 * D), 256, 0, stream>>>(onb, D, w1b, D, x12, 2 * D, nullptr, M, 2 * D, D);
  // hidden = silu(x1) * x2                            (bf16 out)
  k_swiglu<<<(M * D + 255) / 256, 256, 0, stream>>>(x12, hid, M, D);
  // out = hidden @ w3^T + o                           (fp32 out)
  k_gemm<2><<<gblocks(M, D), 256, 0, stream>>>(hid, D, w3b, D, out, D, o, M, D, D);
}
